// H2OOriginal_22883585753581
// MI455X (gfx1250) — matrix operand order repaired and checked by reading, not compiled, not run on hardware
//
#include <hip/hip_runtime.h>
#include <hip/hip_bf16.h>
#include <math.h>

#define SEQ   4096
#define HD    64
#define NH    16
#define BQ    64
#define BK    32
#define SCALE 0.125f

typedef __attribute__((ext_vector_type(16))) _Float16 v16h;
typedef __attribute__((ext_vector_type(8)))  _Float16 v8h;
typedef __attribute__((ext_vector_type(8)))  float    v8f;

union V16u { v16h v; v8h h[2]; };

__device__ __forceinline__ v16h pack2(const _Float16* p0, const _Float16* p1) {
  V16u u;
  u.h[0] = *(const v8h*)p0;
  u.h[1] = *(const v8h*)p1;
  return u.v;
}

__device__ __forceinline__ v8f wmma16(v16h a, v16h b, v8f c) {
  return __builtin_amdgcn_wmma_f32_16x16x32_f16(false, a, false, b, (short)0, c,
                                                false, false);
}

// ---- cross-lane xor-permute on the VALU pipe (v_permlane16_b32) with
// ---- ds_bpermute fallback. Patterns: nibble j of (s0,s1) = j^1 / j^3 / j^7 / j^8 for m = 1 / 2 / 4 / default.
#if __has_builtin(__builtin_amdgcn_permlane16)
__device__ __forceinline__ float xor_lane(float x, int m) {
  int v = __builtin_bit_cast(int, x), r;
  switch (m) {
    case 1: r = __builtin_amdgcn_permlane16(v, v, 0x67452301, 0xEFCDAB89, false, false); break;
    case 2: r = __builtin_amdgcn_permlane16(v, v, 0x45670123, 0xCDEF89AB, false, false); break;
    case 4: r = __builtin_amdgcn_permlane16(v, v, 0x01234567, 0x89ABCDEF, false, false); break;
    default: r = __builtin_amdgcn_permlane16(v, v, 0xFEDCBA98, 0x76543210, false, false); break;
  }
  return __builtin_bit_cast(float, r);
}
#else
__device__ __forceinline__ float xor_lane(float x, int m) {
  return __shfl_xor(x, m, 32);
}
#endif

__device__ __forceinline__ float redmax16(float x) {
#pragma unroll
  for (int m = 1; m < 16; m <<= 1) x = fmaxf(x, xor_lane(x, m));
  return x;
}
__device__ __forceinline__ float redsum16(float x) {
#pragma unroll
  for (int m = 1; m < 16; m <<= 1) x += xor_lane(x, m);
  return x;
}

// ---- CDNA5 async global->LDS copy (ASYNCcnt-tracked, bypasses VGPRs) ----
__device__ __forceinline__ uint32_t lds_addr(const void* p) {
  return (uint32_t)(uintptr_t)p;   // LDS address = low 32 bits of the flat pointer (ISA 10.2)
}
__device__ __forceinline__ void async_b128(uint32_t lds, const _Float16* g) {
  asm volatile("global_load_async_to_lds_b128 %0, %1, off"
               :: "v"(lds), "v"((uint64_t)(uintptr_t)g) : "memory");
}
__device__ __forceinline__ void wait_async0() {
  asm volatile("s_wait_asynccnt 0x0" ::: "memory");
}

// one-time fp32 -> fp16 conversion with K/V re-layout; SCALE folded into Q:
//   qf[h][s][d] = q[h][s][d] * SCALE   (2^-3: exact)
//   kf[h][t][d] = k[h][d][t]           (key tile = contiguous 4KB block)
//   vf[h][d][t] = v[h][t][d]           (V tiles k-contiguous per d row)
__global__ void h2o_cvt_kernel(const float* __restrict__ q,
                               const float* __restrict__ k,
                               const float* __restrict__ v,
                               _Float16* __restrict__ qf,
                               _Float16* __restrict__ kf,
                               _Float16* __restrict__ vf) {
  size_t i = (size_t)blockIdx.x * 256 + threadIdx.x;  // 0 .. NH*SEQ*HD-1
  int d = (int)(i & (HD - 1));
  int s = (int)((i >> 6) & (SEQ - 1));
  int h = (int)(i >> 18);
  qf[i] = (_Float16)(q[i] * SCALE);
  kf[i] = (_Float16)k[((size_t)h * HD + d) * SEQ + s];
  vf[((size_t)h * HD + d) * SEQ + s] = (_Float16)v[i];
}

__global__ void h2o_zero_kernel(float* p, int n) {
  int i = blockIdx.x * blockDim.x + threadIdx.x;
  if (i < n) p[i] = 0.0f;
}

__global__ __launch_bounds__(128) void h2o_attn_kernel(
    const _Float16* __restrict__ qf, const _Float16* __restrict__ kf,
    const _Float16* __restrict__ vf, float* __restrict__ outg,
    float* __restrict__ rsg) {
  __shared__ _Float16 q_lds[BQ * HD];         // [row][d]        8 KB
  __shared__ _Float16 kt_lds[2][BK * HD];     // dbuf [t][d]   2x4 KB
  __shared__ _Float16 v_lds[2][HD * BK];      // dbuf [d][k]   2x4 KB
  __shared__ _Float16 p_lds[4][16 * BK];      // per-wave [row][k] 4 KB

  const int tid  = threadIdx.x;
  const int lane = tid & 31;
  const int wid  = tid >> 5;
  const int g    = lane >> 4;
  const int c    = lane & 15;
  const int h    = blockIdx.y;
  const int qb   = blockIdx.x * BQ;

  const _Float16* qh = qf + ((size_t)h * SEQ + qb) * HD;
  const _Float16* kh = kf + (size_t)h * SEQ * HD;
  const _Float16* vh = vf + (size_t)h * HD * SEQ;
  float* op = outg + (size_t)h * SEQ * HD;
  float* rp = rsg + (size_t)h * SEQ;

  const int nkt = (qb + BQ) / BK;

  auto issue_kv = [&](int t0, int buf) {
    const _Float16* kt = kh + (size_t)t0 * HD;
#pragma unroll
    for (int j = 0; j < 2; ++j) {
      int e = tid + 128 * j;
      async_b128(lds_addr(&kt_lds[buf][0]) + e * 16, kt + e * 8);
      int d = e >> 2, co = (e & 3) * 8;
      async_b128(lds_addr(&v_lds[buf][0]) + (d * BK + co) * 2,
                 vh + (size_t)d * SEQ + t0 + co);
    }
  };
  auto issue_k = [&](int t0, int buf) {
    const _Float16* kt = kh + (size_t)t0 * HD;
#pragma unroll
    for (int j = 0; j < 2; ++j) {
      int e = tid + 128 * j;
      async_b128(lds_addr(&kt_lds[buf][0]) + e * 16, kt + e * 8);
    }
  };
  auto qk_tile = [&](int buf, const v16h& aq0, const v16h& aq1, v8f s[2]) {
#pragma unroll
    for (int tc = 0; tc < 2; ++tc) {
      const _Float16* bcol = &kt_lds[buf][(tc * 16 + c) * HD + 8 * g];
      v16h b0 = pack2(bcol, bcol + 16);
      v16h b1 = pack2(bcol + 32, bcol + 48);
      v8f acc = {};
      acc = wmma16(aq0, b0, acc);
      acc = wmma16(aq1, b1, acc);
      s[tc] = acc;
    }
  };

  // prologue: Q tile + first K/V tile via async copies
#pragma unroll
  for (int j = 0; j < 4; ++j) {
    int e = tid + 128 * j;
    async_b128(lds_addr(&q_lds[0]) + e * 16, qh + e * 8);
  }
  issue_kv(0, 0);
  wait_async0();
  __syncthreads();

  const int qrow0 = qb + wid * 16;
  const _Float16* qrowp = &q_lds[(wid * 16 + c) * HD];
  const v16h aq0 = pack2(qrowp + 0  + 8 * g, qrowp + 16 + 8 * g);
  const v16h aq1 = pack2(qrowp + 32 + 8 * g, qrowp + 48 + 8 * g);

  float mrow[8], lrow[8];
  v8f o[4];
  {
    v8f z = {};
#pragma unroll
    for (int f = 0; f < 4; ++f) o[f] = z;
#pragma unroll
    for (int r = 0; r < 8; ++r) { mrow[r] = -INFINITY; lrow[r] = 0.0f; }
  }

  // ================= pass 1: flash attention =================
  for (int kt = 0; kt < nkt; ++kt) {
    const int t0 = kt * BK;
    const int buf = kt & 1;
    if (kt + 1 < nkt) issue_kv(t0 + BK, buf ^ 1);

    v8f s[2];
    qk_tile(buf, aq0, aq1, s);

    // online softmax; masking only on tiles that touch the causal diagonal
    const bool diag = (t0 + BK - 1 > qrow0);   // wave-uniform
    float p0[8], p1[8], alpha[8];
#pragma unroll
    for (int r = 0; r < 8; ++r) {
      float x0, x1;
      if (diag) {
        const int qrow = qrow0 + 8 * g + r;
        x0 = (t0 + c      <= qrow) ? s[0][r] : -INFINITY;
        x1 = (t0 + 16 + c <= qrow) ? s[1][r] : -INFINITY;
      } else {
        x0 = s[0][r];
        x1 = s[1][r];
      }
      float rm = redmax16(fmaxf(x0, x1));
      float mn = fmaxf(mrow[r], rm);
      float a  = __expf(mrow[r] - mn);
      p0[r] = __expf(x0 - mn);
      p1[r] = __expf(x1 - mn);
      float rs = redsum16(p0[r] + p1[r]);
      lrow[r]  = lrow[r] * a + rs;
      mrow[r]  = mn;
      alpha[r] = a;
    }
#pragma unroll
    for (int r = 0; r < 8; ++r) {
      o[0][r] *= alpha[r]; o[1][r] *= alpha[r];
      o[2][r] *= alpha[r]; o[3][r] *= alpha[r];
    }

    // C-layout -> A-layout transpose of P through wave-private LDS
    _Float16* pw = &p_lds[wid][0];
#pragma unroll
    for (int r = 0; r < 8; ++r) {
      pw[(8 * g + r) * BK + c]      = (_Float16)p0[r];
      pw[(8 * g + r) * BK + 16 + c] = (_Float16)p1[r];
    }
    asm volatile("s_wait_dscnt 0" ::: "memory");
    v16h ap = pack2(&pw[c * BK + 8 * g], &pw[c * BK + 16 + 8 * g]);

    // PV: O += P(16x32) * V(32x64)
#pragma unroll
    for (int f = 0; f < 4; ++f) {
      const _Float16* bv = &v_lds[buf][(16 * f + c) * BK + 8 * g];
      v16h b = pack2(bv, bv + 16);
      o[f] = wmma16(ap, b, o[f]);
    }

    wait_async0();
    __syncthreads();
  }

  // write out = O / l
#pragma unroll
  for (int r = 0; r < 8; ++r) {
    float inv = 1.0f / lrow[r];
    int row = qrow0 + 8 * g + r;
#pragma unroll
    for (int f = 0; f < 4; ++f)
      op[(size_t)row * HD + 16 * f + c] = o[f][r] * inv;
  }

  // ================= pass 2: per-key accumulated attention =================
  float madj[8];
#pragma unroll
  for (int r = 0; r < 8; ++r) madj[r] = mrow[r] + __logf(lrow[r]);

  __syncthreads();
  issue_k(0, 0);
  for (int kt = 0; kt < nkt; ++kt) {
    const int t0 = kt * BK;
    const int buf = kt & 1;
    wait_async0();
    __syncthreads();
    if (kt + 1 < nkt) issue_k(t0 + BK, buf ^ 1);

    v8f s[2];
    qk_tile(buf, aq0, aq1, s);

    const bool diag = (t0 + BK - 1 > qrow0);
    float cs0 = 0.0f, cs1 = 0.0f;
#pragma unroll
    for (int r = 0; r < 8; ++r) {
      float x0, x1;
      if (diag) {
        const int qrow = qrow0 + 8 * g + r;
        x0 = (t0 + c      <= qrow) ? s[0][r] : -INFINITY;
        x1 = (t0 + 16 + c <= qrow) ? s[1][r] : -INFINITY;
      } else {
        x0 = s[0][r];
        x1 = s[1][r];
      }
      cs0 += __expf(x0 - madj[r]);   // normalized p = exp(s - m - log l)
      cs1 += __expf(x1 - madj[r]);
    }
    cs0 += xor_lane(cs0, 16);
    cs1 += xor_lane(cs1, 16);
    float addv = (g == 0) ? cs0 : cs1;
    int col = t0 + 16 * g + c;
    atomicAdd(&rp[col], addv);
  }
}

extern "C" void kernel_launch(void* const* d_in, const int* in_sizes, int n_in,
                              void* d_out, int out_size, void* d_ws,
                              size_t ws_size, hipStream_t stream) {
  (void)in_sizes; (void)n_in; (void)ws_size; (void)out_size;
  const float* q = (const float*)d_in[0];
  const float* k = (const float*)d_in[1];   // [B,H,D,S] pre-transposed
  const float* v = (const float*)d_in[2];
  float* out = (float*)d_out;                               // [B,H,S,D]
  float* rs  = out + (size_t)1 * NH * SEQ * HD;             // [B,H,S]

  const size_t m = (size_t)NH * SEQ * HD;
  _Float16* qf = (_Float16*)d_ws;           // 24 MB of workspace
  _Float16* kf = qf + m;
  _Float16* vf = kf + m;

  h2o_cvt_kernel<<<(unsigned)(m / 256), 256, 0, stream>>>(q, k, v, qf, kf, vf);
  h2o_zero_kernel<<<(NH * SEQ + 255) / 256, 256, 0, stream>>>(rs, NH * SEQ);
  dim3 grid(SEQ / BQ, NH);
  h2o_attn_kernel<<<grid, 128, 0, stream>>>(qf, kf, vf, out, rs);
}
